// ModalTail_850403525410
// MI455X (gfx1250) — compile-verified, matrix operand order checked by reading, not run on hardware
//
#include <hip/hip_runtime.h>
#include <hip/hip_bf16.h>
#include <math.h>

// ---------------- problem constants (from reference) ----------------
#define B_    2
#define N_    512
#define D_    512
#define F_    64
#define O_    16
#define H_    1024
#define H2_   2048
#define DCTX_ 128
#define NF_   256

#define MT    64      // rows (n-values) per block tile
#define LDN   520     // sN row stride in bf16 (1040B, 16B multiple)
#define LDS_S 1032    // sS row stride in bf16 (2064B, 16B multiple)

typedef __attribute__((ext_vector_type(16))) __bf16 v16bf;
typedef __attribute__((ext_vector_type(8)))  __bf16 v8bf;
typedef __attribute__((ext_vector_type(8)))  float  v8f;

#define WMMA_BF16(a, b, c) \
  __builtin_amdgcn_wmma_f32_16x16x32_bf16(false, (a), false, (b), (short)0, (c), false, false)

// Load a 16-element bf16 fragment: two contiguous 16B chunks (K..K+7, K+16..K+23)
__device__ __forceinline__ v16bf frag2(const __bf16* p) {
  union { v16bf v; v8bf h[2]; } u;
  u.h[0] = *(const v8bf*)(p);
  u.h[1] = *(const v8bf*)(p + 16);
  return u.v;
}

// A-matrix fragment (16x32 MxK) from row-major bf16 (stride in elements).
// Lane L: m = L&15, kbase = (L>>4)*8; elems 0..7 = K kbase..+7, 8..15 = K kbase+16..+23
__device__ __forceinline__ v16bf load_a(const __bf16* base, int stride, int m0, int k0, int lane) {
  int m  = m0 + (lane & 15);
  int kb = k0 + ((lane >> 4) << 3);
  return frag2(base + (size_t)m * stride + kb);
}

// B-matrix fragment (32x16 KxN) from N-major bf16 weight (row n holds K contiguous).
__device__ __forceinline__ v16bf load_b(const __bf16* wt, int Kdim, int n0, int k0, int lane) {
  int n  = n0 + (lane & 15);
  int kb = k0 + ((lane >> 4) << 3);
  return frag2(wt + (size_t)n * Kdim + kb);
}

// ---------------- prep kernels ----------------

// ss[b, 2D] = ctx[b] @ ln_w + ln_b   (tiny: 2 x 1024 outputs, K=128)
__global__ void k_ss(const float* __restrict__ ctx, const float* __restrict__ ln_w,
                     const float* __restrict__ ln_b, float* __restrict__ ss) {
  int b = blockIdx.x;
  for (int j = threadIdx.x; j < 2 * D_; j += 256) {
    float acc = ln_b[j];
    #pragma unroll 4
    for (int k = 0; k < DCTX_; ++k) acc += ctx[b * DCTX_ + k] * ln_w[k * (2 * D_) + j];
    ss[b * 2 * D_ + j] = acc;
  }
}

// wt_in[n, k] = bf16(ffn_w_in[k, n]);  n < 2048, k < 512
__global__ void k_t_win(const float* __restrict__ w, __bf16* __restrict__ wt) {
  int i = blockIdx.x * 256 + threadIdx.x;
  if (i >= H2_ * D_) return;
  int n = i / D_, k = i % D_;
  wt[i] = (__bf16)w[k * H2_ + n];
}

// wt_out[d, h] = bf16(ffn_w_out[h, d]); d < 512, h < 1024
__global__ void k_t_wout(const float* __restrict__ w, __bf16* __restrict__ wt) {
  int i = blockIdx.x * 256 + threadIdx.x;
  if (i >= D_ * H_) return;
  int d = i / H_, h = i % H_;
  wt[i] = (__bf16)w[h * D_ + d];
}

// wt_proj[feat, o, d] = bf16(out_proj[feat, d*16 + o])
__global__ void k_t_wproj(const float* __restrict__ w, __bf16* __restrict__ wt) {
  int i = blockIdx.x * 256 + threadIdx.x;
  if (i >= NF_ * O_ * D_) return;
  int feat = i / (O_ * D_);
  int o    = (i / D_) % O_;
  int d    = i % D_;
  wt[i] = (__bf16)w[(size_t)feat * (D_ * O_) + d * O_ + o];
}

// ---------------- fused main kernel ----------------
extern __shared__ char smem_raw[];

__global__ __launch_bounds__(256, 1)
void k_fused(const float* __restrict__ x, const int* __restrict__ idx,
             const float* __restrict__ feature_bias,
             const float* __restrict__ ffn_b_in, const float* __restrict__ ffn_b_out,
             const float* __restrict__ ss,
             const __bf16* __restrict__ wt_in,   // (2048, 512)
             const __bf16* __restrict__ wt_out,  // (512, 1024)
             const __bf16* __restrict__ wt_proj, // (256, 16, 512)
             float* __restrict__ out)            // (B, N, F, O)
{
  __bf16* sN = (__bf16*)smem_raw;          // 64 x LDN   : norm rows, later xr rows
  __bf16* sS = sN + MT * LDN;              // 64 x LDS_S : silu(g)*v rows

  const int bid   = blockIdx.x;
  const int ntile = bid & 7;               // N_/MT = 8
  const int f     = (bid >> 3) & (F_ - 1);
  const int b     = bid >> 9;              // / (8*64)
  const int n0    = ntile * MT;

  const int lane = threadIdx.x & 31;
  const int wave = threadIdx.x >> 5;

  const int feat = idx[b * F_ + f];
  const float* bias = feature_bias + (size_t)feat * D_;
  const float* scp  = ss + (size_t)b * (2 * D_);
  const float* shp  = scp + D_;

  // ---- Stage A: h = x + bias; LayerNorm; AdaLN modulate; bf16 -> sN ----
  for (int r = wave; r < MT; r += 8) {
    const float* xrow = x + ((size_t)b * N_ + (n0 + r)) * D_;
    const int base = lane * 16;
    float v[16];
    float s1 = 0.f, s2 = 0.f;
    #pragma unroll
    for (int i = 0; i < 16; i += 4) {
      float4 xv = *(const float4*)(xrow + base + i);
      float4 bv = *(const float4*)(bias + base + i);
      v[i+0] = xv.x + bv.x; v[i+1] = xv.y + bv.y;
      v[i+2] = xv.z + bv.z; v[i+3] = xv.w + bv.w;
      s1 += v[i+0] + v[i+1] + v[i+2] + v[i+3];
      s2 += v[i+0]*v[i+0] + v[i+1]*v[i+1] + v[i+2]*v[i+2] + v[i+3]*v[i+3];
    }
    #pragma unroll
    for (int off = 16; off > 0; off >>= 1) {
      s1 += __shfl_xor(s1, off, 32);
      s2 += __shfl_xor(s2, off, 32);
    }
    const float mu  = s1 * (1.0f / D_);
    const float var = s2 * (1.0f / D_) - mu * mu;
    const float inv = rsqrtf(var + 1e-5f);
    #pragma unroll
    for (int i = 0; i < 16; ++i) {
      const int d = base + i;
      float nv = (v[i] - mu) * inv;
      nv = nv * (1.0f + scp[d]) + shp[d];
      sN[r * LDN + d] = (__bf16)nv;
    }
  }
  __syncthreads();

  // ---- Stage B: u = norm @ W_in ; s = silu(g) * v -> sS (bf16) ----
  // wave w owns hidden (gate) columns [w*128, w*128+128)
  {
    const int colbase = wave * 128;
    for (int c = 0; c < 4; ++c) {
      const int cb = colbase + c * 32;
      v8f accg[4][2], accv[4][2];
      #pragma unroll
      for (int mi = 0; mi < 4; ++mi)
        #pragma unroll
        for (int nj = 0; nj < 2; ++nj) { accg[mi][nj] = (v8f){}; accv[mi][nj] = (v8f){}; }

      for (int k = 0; k < D_; k += 32) {
        if (k + 32 < D_) {
          __builtin_prefetch(wt_in + (size_t)(cb + (lane & 15)) * D_ + k + 32, 0, 1);
          __builtin_prefetch(wt_in + (size_t)(H_ + cb + (lane & 15)) * D_ + k + 32, 0, 1);
        }
        v16bf a[4];
        #pragma unroll
        for (int mi = 0; mi < 4; ++mi) a[mi] = load_a(sN, LDN, mi * 16, k, lane);
        #pragma unroll
        for (int nj = 0; nj < 2; ++nj) {
          v16bf bg = load_b(wt_in, D_, cb + nj * 16, k, lane);
          v16bf bv = load_b(wt_in, D_, H_ + cb + nj * 16, k, lane);
          #pragma unroll
          for (int mi = 0; mi < 4; ++mi) {
            accg[mi][nj] = WMMA_BF16(a[mi], bg, accg[mi][nj]);
            accv[mi][nj] = WMMA_BF16(a[mi], bv, accv[mi][nj]);
          }
        }
      }
      // fused bias + SiLU gate, write bf16 tile to sS
      #pragma unroll
      for (int nj = 0; nj < 2; ++nj) {
        const int col = cb + nj * 16 + (lane & 15);
        const float gb = ffn_b_in[col];
        const float vb = ffn_b_in[H_ + col];
        #pragma unroll
        for (int mi = 0; mi < 4; ++mi) {
          #pragma unroll
          for (int r = 0; r < 8; ++r) {
            const float g  = accg[mi][nj][r] + gb;
            const float vv = accv[mi][nj][r] + vb;
            const float sg = g / (1.0f + __expf(-g));
            const int row  = mi * 16 + r + ((lane >> 4) << 3);
            sS[row * LDS_S + col] = (__bf16)(sg * vv);
          }
        }
      }
    }
  }
  __syncthreads();

  // ---- Stage C: fout = s @ W_out + b_out ; xr = x + fout -> sN (bf16) ----
  // wave w owns output columns [w*64, w*64+64)
  {
    const int ob = wave * 64;
    v8f acc[4][4];
    #pragma unroll
    for (int mi = 0; mi < 4; ++mi)
      #pragma unroll
      for (int nj = 0; nj < 4; ++nj) acc[mi][nj] = (v8f){};

    for (int k = 0; k < H_; k += 32) {
      if (k + 32 < H_)
        __builtin_prefetch(wt_out + (size_t)(ob + (lane & 15)) * H_ + k + 32, 0, 1);
      v16bf a[4];
      #pragma unroll
      for (int mi = 0; mi < 4; ++mi) a[mi] = load_a(sS, LDS_S, mi * 16, k, lane);
      #pragma unroll
      for (int nj = 0; nj < 4; ++nj) {
        v16bf bw = load_b(wt_out, H_, ob + nj * 16, k, lane);
        #pragma unroll
        for (int mi = 0; mi < 4; ++mi) acc[mi][nj] = WMMA_BF16(a[mi], bw, acc[mi][nj]);
      }
    }
    #pragma unroll
    for (int nj = 0; nj < 4; ++nj) {
      const int col = ob + nj * 16 + (lane & 15);
      const float bo = ffn_b_out[col];
      #pragma unroll
      for (int mi = 0; mi < 4; ++mi) {
        #pragma unroll
        for (int r = 0; r < 8; ++r) {
          const int row = mi * 16 + r + ((lane >> 4) << 3);
          const float xr = acc[mi][nj][r] + bo + x[((size_t)b * N_ + n0 + row) * D_ + col];
          sN[row * LDN + col] = (__bf16)xr;   // sN reused: norm dead after stage B
        }
      }
    }
  }
  __syncthreads();

  // ---- Stage D: out[64,16] = xr @ w_proj(512x16) ----
  if (wave < 4) {
    const __bf16* wp = wt_proj + (size_t)feat * (O_ * D_); // (16, 512) o-major
    const int m0 = wave * 16;
    v8f acc = (v8f){};
    for (int k = 0; k < D_; k += 32) {
      v16bf a  = load_a(sN, LDN, m0, k, lane);
      v16bf bw = load_b(wp, D_, 0, k, lane);
      acc = WMMA_BF16(a, bw, acc);
    }
    const int o = lane & 15;
    #pragma unroll
    for (int r = 0; r < 8; ++r) {
      const int row = m0 + r + ((lane >> 4) << 3);
      out[(((size_t)b * N_ + (n0 + row)) * F_ + f) * O_ + o] = acc[r];
    }
  }
}

// ---------------- host launcher ----------------
extern "C" void kernel_launch(void* const* d_in, const int* in_sizes, int n_in,
                              void* d_out, int out_size, void* d_ws, size_t ws_size,
                              hipStream_t stream) {
  (void)in_sizes; (void)n_in; (void)out_size; (void)ws_size;

  const float* x            = (const float*)d_in[0];
  const int*   idx          = (const int*)d_in[1];
  const float* ctx          = (const float*)d_in[2];
  const float* feature_bias = (const float*)d_in[3];
  const float* out_proj     = (const float*)d_in[4];
  const float* ln_w         = (const float*)d_in[5];
  const float* ln_b         = (const float*)d_in[6];
  const float* ffn_w_in     = (const float*)d_in[7];
  const float* ffn_b_in     = (const float*)d_in[8];
  const float* ffn_w_out    = (const float*)d_in[9];
  const float* ffn_b_out    = (const float*)d_in[10];
  float* out = (float*)d_out;

  char* ws = (char*)d_ws;
  float*  ss      = (float*)ws;                               // 8 KB
  __bf16* wt_in   = (__bf16*)(ws + 8192);                     // 2 MB
  __bf16* wt_out  = wt_in  + (size_t)H2_ * D_;                // 1 MB
  __bf16* wt_proj = wt_out + (size_t)D_ * H_;                 // 4 MB

  k_ss     <<<B_, 256, 0, stream>>>(ctx, ln_w, ln_b, ss);
  k_t_win  <<<(H2_ * D_) / 256, 256, 0, stream>>>(ffn_w_in,  wt_in);
  k_t_wout <<<(D_ * H_)  / 256, 256, 0, stream>>>(ffn_w_out, wt_out);
  k_t_wproj<<<(NF_ * O_ * D_) / 256, 256, 0, stream>>>(out_proj, wt_proj);

  const size_t smem = (size_t)(MT * LDN + MT * LDS_S) * sizeof(__bf16); // ~194 KB
  dim3 grid(B_ * F_ * (N_ / MT));   // 1024 blocks
  k_fused<<<grid, 256, smem, stream>>>(x, idx, feature_bias, ffn_b_in, ffn_b_out,
                                       ss, wt_in, wt_out, wt_proj, out);
}
